// HEATLayer_23957327577905
// MI455X (gfx1250) — matrix operand order repaired and checked by reading, not compiled, not run on hardware
//
#include <hip/hip_runtime.h>
#include <hip/hip_bf16.h>

typedef __attribute__((ext_vector_type(16))) _Float16 v16h;
typedef __attribute__((ext_vector_type(8)))  _Float16 v8h;
typedef __attribute__((ext_vector_type(4)))  _Float16 v4h;
typedef __attribute__((ext_vector_type(8)))  float    v8f;

#define N_NODES 50000
#define E_EDGES 800000
#define FEAT    128
#define HEADS   8
#define DK      16
#define MTILES  3125   // 50000 / 16

// ---------- order-preserving float <-> uint encoding for atomic max ----------
__device__ __forceinline__ unsigned enc_f(float f) {
  unsigned u = __float_as_uint(f);
  return (u & 0x80000000u) ? ~u : (u | 0x80000000u);
}
__device__ __forceinline__ float dec_f(unsigned e) {
  unsigned u = (e & 0x80000000u) ? (e ^ 0x80000000u) : ~e;
  return __uint_as_float(u);
}

// ---------- utility kernels ----------
__global__ void fill_u32(unsigned* __restrict__ p, unsigned v, int n) {
  int i = blockIdx.x * blockDim.x + threadIdx.x;
  if (i < n) p[i] = v;
}

// float -> f16 feature conversion, 4 elems/thread
__global__ void cvt_half4(const float* __restrict__ in, _Float16* __restrict__ out, int n4) {
  int i = blockIdx.x * blockDim.x + threadIdx.x;
  if (i >= n4) return;
  float4 f = ((const float4*)in)[i];
  v4h h; h[0] = (_Float16)f.x; h[1] = (_Float16)f.y; h[2] = (_Float16)f.z; h[3] = (_Float16)f.w;
  ((v4h*)out)[i] = h;
}

// Repack a 128x128 row-major f32 weight into WMMA B-fragment order (f16).
// frag[((ntile*4+kk)*32+lane)*16 + j]:
//   lane<16 : n = ntile*16+lane,     k = kk*32 + (j<8 ? j : 16 + j-8)
//   lane>=16: n = ntile*16+lane-16,  k = kk*32 + (j<8 ? 8 + j : 24 + j-8)
__global__ void repack_w(const float* __restrict__ w, _Float16* __restrict__ frag) {
  int idx = blockIdx.x * 256 + threadIdx.x;          // 0..16383
  int j     = idx & 15;
  int lane  = (idx >> 4) & 31;
  int kk    = (idx >> 9) & 3;
  int ntile = idx >> 11;
  int n = ntile * 16 + (lane & 15);
  int k = kk * 32 + ((lane < 16) ? 0 : 8) + ((j < 8) ? j : 8 + j);
  frag[idx] = (_Float16)w[k * 128 + n];
}

// ---------- WMMA GEMM: Out16[M,128] = f16( A16[M,128] @ W + bias ) ----------
__global__ void gemm_qkv(const _Float16* __restrict__ A16,
                         const _Float16* __restrict__ Wfrag,
                         const float* __restrict__ bias,
                         _Float16* __restrict__ Out16) {
  __shared__ _Float16 ldsA[16 * 128];
  const int tile = blockIdx.x;
  const int t    = threadIdx.x;
  { // stage 16x128 f16 A tile: 16 threads per row, 8 halves each
    int row = t >> 4, seg = t & 15;
    *(v8h*)(ldsA + row * 128 + seg * 8) =
        *(const v8h*)(A16 + (size_t)(tile * 16 + row) * 128 + seg * 8);
  }
  __syncthreads();
  const int wave = t >> 5;           // N-tile 0..7
  const int lane = t & 31;
  const bool lo  = lane < 16;
  const int row  = lane & 15;
  v8f c = {};
#pragma unroll
  for (int kk = 0; kk < 4; ++kk) {
    int b0 = kk * 32 + (lo ? 0 : 8);
    int b1 = kk * 32 + (lo ? 16 : 24);
    v8h a0 = *(const v8h*)(ldsA + row * 128 + b0);   // ds_load_b128
    v8h a1 = *(const v8h*)(ldsA + row * 128 + b1);   // ds_load_b128
    v16h a;
#pragma unroll
    for (int j = 0; j < 8; ++j) { a[j] = a0[j]; a[8 + j] = a1[j]; }
    v16h b = *(const v16h*)(Wfrag + (size_t)(((wave << 2) + kk) * 32 + lane) * 16);
    c = __builtin_amdgcn_wmma_f32_16x16x32_f16(false, a, false, b, (short)0, c, false, false);
  }
  int col = wave * 16 + row;
  float bb = bias[col];
#pragma unroll
  for (int r = 0; r < 8; ++r) {
    int m = lo ? r : 8 + r;
    Out16[(size_t)(tile * 16 + m) * 128 + col] = (_Float16)(c[r] + bb);
  }
}

// ---------- output GEMM + skip blend: out = (scale*agg @ Aw + Ab)*alpha + feat*(1-alpha)
__global__ void out_gemm(const float* __restrict__ Agg, float scale,
                         const _Float16* __restrict__ Wfrag,
                         const float* __restrict__ bias,
                         const float* __restrict__ skipArr, int nid,
                         const float* __restrict__ feat,
                         float* __restrict__ Out) {
  __shared__ _Float16 ldsA[16 * 128];
  const int tile = blockIdx.x;
  const int t    = threadIdx.x;
  {
    int row = t >> 4, seg = t & 15;
    const float* src = Agg + (size_t)(tile * 16 + row) * 128 + seg * 8;
    _Float16* dst = ldsA + row * 128 + seg * 8;
#pragma unroll
    for (int j = 0; j < 8; ++j) dst[j] = (_Float16)(src[j] * scale);
  }
  __syncthreads();
  const int wave = t >> 5;
  const int lane = t & 31;
  const bool lo  = lane < 16;
  const int row  = lane & 15;
  v8f c = {};
#pragma unroll
  for (int kk = 0; kk < 4; ++kk) {
    int b0 = kk * 32 + (lo ? 0 : 8);
    int b1 = kk * 32 + (lo ? 16 : 24);
    v8h a0 = *(const v8h*)(ldsA + row * 128 + b0);
    v8h a1 = *(const v8h*)(ldsA + row * 128 + b1);
    v16h a;
#pragma unroll
    for (int j = 0; j < 8; ++j) { a[j] = a0[j]; a[8 + j] = a1[j]; }
    v16h b = *(const v16h*)(Wfrag + (size_t)(((wave << 2) + kk) * 32 + lane) * 16);
    c = __builtin_amdgcn_wmma_f32_16x16x32_f16(false, a, false, b, (short)0, c, false, false);
  }
  float alpha = 1.f / (1.f + __expf(-skipArr[nid]));
  float beta  = 1.f - alpha;
  int col = wave * 16 + row;
  float bb = bias[col];
#pragma unroll
  for (int r = 0; r < 8; ++r) {
    int m = lo ? r : 8 + r;
    size_t idx = (size_t)(tile * 16 + m) * 128 + col;
    Out[idx] = (c[r] + bb) * alpha + feat[idx] * beta;
  }
}

// ---------- edge pass A: score + segmented atomic max ----------
__global__ void edge_score(const _Float16* __restrict__ q16, const _Float16* __restrict__ k16,
                           const int* __restrict__ src, const int* __restrict__ dst,
                           const float* __restrict__ ev, const float* __restrict__ ewp,
                           const float* __restrict__ ebp, unsigned* __restrict__ menc,
                           float* __restrict__ score) {
  int tid = blockIdx.x * blockDim.x + threadIdx.x;
  if (tid >= E_EDGES * HEADS) return;
  int e = tid >> 3, h = tid & 7;
  int s = src[e], d = dst[e];
  v16h qv = *(const v16h*)(q16 + (size_t)d * 128 + h * 16);   // 32B gather
  v16h kv = *(const v16h*)(k16 + (size_t)s * 128 + h * 16);   // 32B gather
  float acc = 0.f;
#pragma unroll
  for (int j = 0; j < 16; ++j) acc += (float)qv[j] * (float)kv[j];
  float ea = ev[e] * ewp[0] + ebp[0];
  float sc = acc * ea * 0.25f;                 // 1/sqrt(DK), DK=16
  score[tid] = sc;
  atomicMax(menc + (size_t)d * 8 + h, enc_f(sc));
}

// ---------- edge pass B: exp(score - max) + segmented sum ----------
__global__ void edge_exp(float* __restrict__ score, const int* __restrict__ dst,
                         const unsigned* __restrict__ menc, float* __restrict__ z) {
  int tid = blockIdx.x * blockDim.x + threadIdx.x;
  if (tid >= E_EDGES * HEADS) return;
  int e = tid >> 3, h = tid & 7;
  int d = dst[e];
  float m  = dec_f(menc[(size_t)d * 8 + h]);
  float ex = __expf(score[tid] - m);
  score[tid] = ex;
  atomicAdd(z + (size_t)d * 8 + h, ex);
}

// ---------- edge pass C: agg[dst] += v[src] * attn ----------
__global__ void edge_agg(const float* __restrict__ score, const float* __restrict__ z,
                         const _Float16* __restrict__ v16p, const int* __restrict__ src,
                         const int* __restrict__ dst, float* __restrict__ agg) {
  int tid = blockIdx.x * blockDim.x + threadIdx.x;
  if (tid >= E_EDGES * 32) return;
  int e = tid >> 5, l = tid & 31;
  int h = l >> 2;
  int s = src[e], d = dst[e];
  float attn = score[(size_t)e * 8 + h] / z[(size_t)d * 8 + h];
  v4h vv = *(const v4h*)(v16p + (size_t)s * 128 + l * 4);
  float* ap = agg + (size_t)d * 128 + l * 4;
#pragma unroll
  for (int j = 0; j < 4; ++j) atomicAdd(ap + j, (float)vv[j] * attn);
}

// ==========================================================================
extern "C" void kernel_launch(void* const* d_in, const int* in_sizes, int n_in,
                              void* d_out, int out_size, void* d_ws, size_t ws_size,
                              hipStream_t stream) {
  const float* feat0 = (const float*)d_in[0];
  const float* feat1 = (const float*)d_in[1];
  const float* ev0   = (const float*)d_in[2];
  const float* ev1   = (const float*)d_in[3];
  const float* ev2   = (const float*)d_in[4];
  const int*   src0  = (const int*)d_in[5];
  const int*   dst0  = (const int*)d_in[6];
  const int*   src1  = (const int*)d_in[7];
  const int*   dst1  = (const int*)d_in[8];
  const int*   src2  = (const int*)d_in[9];
  const int*   dst2  = (const int*)d_in[10];
  const float* Kw = (const float*)d_in[11];
  const float* Kb = (const float*)d_in[12];
  const float* Qw = (const float*)d_in[13];
  const float* Qb = (const float*)d_in[14];
  const float* Vw = (const float*)d_in[15];
  const float* Vb = (const float*)d_in[16];
  const float* Aw = (const float*)d_in[17];
  const float* Ab = (const float*)d_in[18];
  const float* ew = (const float*)d_in[19];
  const float* eb = (const float*)d_in[20];
  const float* skip = (const float*)d_in[21];

  const size_t NF = (size_t)N_NODES * FEAT;      // 6.4M elems
  char* ws = (char*)d_ws;
  size_t off = 0;
  auto take = [&](size_t bytes) -> char* {
    char* p = ws + off;
    off += (bytes + 255) & ~(size_t)255;
    return p;
  };
  _Float16* f16_0 = (_Float16*)take(NF * 2);
  _Float16* f16_1 = (_Float16*)take(NF * 2);
  _Float16* k0 = (_Float16*)take(NF * 2);
  _Float16* q0 = (_Float16*)take(NF * 2);
  _Float16* v0 = (_Float16*)take(NF * 2);
  _Float16* k1 = (_Float16*)take(NF * 2);
  _Float16* q1 = (_Float16*)take(NF * 2);
  _Float16* v1 = (_Float16*)take(NF * 2);
  _Float16* wf = (_Float16*)take(8 * 16384 * 2);   // K0 K1 Q0 Q1 V0 V1 A0 A1
  float*    agg0 = (float*)take(NF * 4);
  float*    agg1 = (float*)take(NF * 4);
  float*    score = (float*)take((size_t)E_EDGES * 8 * 4);
  unsigned* menc = (unsigned*)take((size_t)N_NODES * 8 * 4);
  float*    zbuf = (float*)take((size_t)N_NODES * 8 * 4);

  _Float16* wfK0 = wf;               _Float16* wfK1 = wf + 16384;
  _Float16* wfQ0 = wf + 2 * 16384;   _Float16* wfQ1 = wf + 3 * 16384;
  _Float16* wfV0 = wf + 4 * 16384;   _Float16* wfV1 = wf + 5 * 16384;
  _Float16* wfA0 = wf + 6 * 16384;   _Float16* wfA1 = wf + 7 * 16384;

  const int NH = N_NODES * HEADS;                 // 400000
  const int gNH = (NH + 255) / 256;
  const int gEH = (E_EDGES * HEADS + 255) / 256;  // 25000
  const int gEL = (E_EDGES * 32) / 256;           // 100000
  const int gNF4 = (int)(NF / 4 + 255) / 256;

  // 1) f32 -> f16 features
  cvt_half4<<<gNF4, 256, 0, stream>>>(feat0, f16_0, (int)(NF / 4));
  cvt_half4<<<gNF4, 256, 0, stream>>>(feat1, f16_1, (int)(NF / 4));

  // 2) repack the 8 weight matrices into WMMA B-fragment order
  repack_w<<<64, 256, 0, stream>>>(Kw,          wfK0);
  repack_w<<<64, 256, 0, stream>>>(Kw + 16384,  wfK1);
  repack_w<<<64, 256, 0, stream>>>(Qw,          wfQ0);
  repack_w<<<64, 256, 0, stream>>>(Qw + 16384,  wfQ1);
  repack_w<<<64, 256, 0, stream>>>(Vw,          wfV0);
  repack_w<<<64, 256, 0, stream>>>(Vw + 16384,  wfV1);
  repack_w<<<64, 256, 0, stream>>>(Aw,          wfA0);
  repack_w<<<64, 256, 0, stream>>>(Aw + 16384,  wfA1);

  // 3) K/Q/V projections (WMMA)
  gemm_qkv<<<MTILES, 256, 0, stream>>>(f16_0, wfK0, Kb,       k0);
  gemm_qkv<<<MTILES, 256, 0, stream>>>(f16_0, wfQ0, Qb,       q0);
  gemm_qkv<<<MTILES, 256, 0, stream>>>(f16_0, wfV0, Vb,       v0);
  gemm_qkv<<<MTILES, 256, 0, stream>>>(f16_1, wfK1, Kb + 128, k1);
  gemm_qkv<<<MTILES, 256, 0, stream>>>(f16_1, wfQ1, Qb + 128, q1);
  gemm_qkv<<<MTILES, 256, 0, stream>>>(f16_1, wfV1, Vb + 128, v1);

  // 4) zero the aggregation buffers
  fill_u32<<<(int)(NF + 255) / 256, 256, 0, stream>>>((unsigned*)agg0, 0u, (int)NF);
  fill_u32<<<(int)(NF + 255) / 256, 256, 0, stream>>>((unsigned*)agg1, 0u, (int)NF);

  // 5) etype0: t0 -> t1  (k/v from type0, q from type1) -> agg1
  fill_u32<<<gNH, 256, 0, stream>>>(menc, 0u, NH);
  fill_u32<<<gNH, 256, 0, stream>>>((unsigned*)zbuf, 0u, NH);
  edge_score<<<gEH, 256, 0, stream>>>(q1, k0, src0, dst0, ev0, ew, eb, menc, score);
  edge_exp  <<<gEH, 256, 0, stream>>>(score, dst0, menc, zbuf);
  edge_agg  <<<gEL, 256, 0, stream>>>(score, zbuf, v0, src0, dst0, agg1);

  // 6) etype1: t1 -> t0 -> agg0
  fill_u32<<<gNH, 256, 0, stream>>>(menc, 0u, NH);
  fill_u32<<<gNH, 256, 0, stream>>>((unsigned*)zbuf, 0u, NH);
  edge_score<<<gEH, 256, 0, stream>>>(q0, k1, src1, dst1, ev1, ew, eb, menc, score);
  edge_exp  <<<gEH, 256, 0, stream>>>(score, dst1, menc, zbuf);
  edge_agg  <<<gEL, 256, 0, stream>>>(score, zbuf, v1, src1, dst1, agg0);

  // 7) etype2: t0 -> t0 -> agg0
  fill_u32<<<gNH, 256, 0, stream>>>(menc, 0u, NH);
  fill_u32<<<gNH, 256, 0, stream>>>((unsigned*)zbuf, 0u, NH);
  edge_score<<<gEH, 256, 0, stream>>>(q0, k0, src2, dst2, ev2, ew, eb, menc, score);
  edge_exp  <<<gEH, 256, 0, stream>>>(score, dst2, menc, zbuf);
  edge_agg  <<<gEL, 256, 0, stream>>>(score, zbuf, v0, src2, dst2, agg0);

  // 8) output projection + skip blend; mean over the two etypes into t0 -> scale 0.5
  float* out = (float*)d_out;
  out_gemm<<<MTILES, 256, 0, stream>>>(agg0, 0.5f, wfA0, Ab,       skip, 0, feat0, out);
  out_gemm<<<MTILES, 256, 0, stream>>>(agg1, 1.0f, wfA1, Ab + 128, skip, 1, feat1, out + NF);
}
